// SDFModel_90675349553405
// MI455X (gfx1250) — matrix operand order repaired and checked by reading, not compiled, not run on hardware
//
#include <hip/hip_runtime.h>
#include <hip/hip_bf16.h>

// ---------------------------------------------------------------------------
// SDF model, MI455X (gfx1250, wave32, WMMA).
//
// Shapes: T=256, N=4000, DM=178, DI=46, H=64, DH=64, F=DI+H=110 (pad->128).
// Heavy part: pointwise MLP over T*N=1,024,000 rows: 110->64->64->1
// (~23 GFLOP) fused into one WMMA kernel (f16 in, f32 accumulate).
// LSTM: parallel input projection + one sequential 256-step block.
// Final reduction: deterministic (no float atomics).
// ---------------------------------------------------------------------------

typedef _Float16 v16h __attribute__((ext_vector_type(16)));
typedef _Float16 h8   __attribute__((ext_vector_type(8)));
typedef float    v8f  __attribute__((ext_vector_type(8)));

#define TT   256
#define NN   4000
#define DM_  178
#define DI_  46
#define HH   64
#define FP   128   // F=110 padded to 128 (4 K-steps of 32)
#define DH_  64

static __device__ inline v16h cat16(h8 lo, h8 hi) {
    return __builtin_shufflevector(lo, hi, 0,1,2,3,4,5,6,7,8,9,10,11,12,13,14,15);
}

// ---------------------------------------------------------------------------
// Kernel A1: xz[t, j] = (b_ih+b_hh)[j] + sum_k macro[t,k] * W_ih[j,k]
// grid = T blocks, 256 threads (j = 0..4H-1). Fully parallel, tiny.
// ---------------------------------------------------------------------------
__global__ __launch_bounds__(256) void lstm_xz_kernel(
    const float* __restrict__ macro, const float* __restrict__ W_ih,
    const float* __restrict__ b_ih, const float* __restrict__ b_hh,
    float* __restrict__ xz)
{
    const int j = threadIdx.x;           // 0..255 (4H)
    const int t = blockIdx.x;            // 0..255
    const float* xr = macro + t * DM_;
    const float* wr = W_ih + j * DM_;
    float acc = b_ih[j] + b_hh[j];
    #pragma unroll 2
    for (int k = 0; k < DM_; ++k) acc = fmaf(xr[k], wr[k], acc);
    xz[t * 256 + j] = acc;
}

// ---------------------------------------------------------------------------
// Kernel A2: sequential LSTM recurrence. One block, 256 threads.
// Thread j owns gate-row j of W_hh (64 f32 kept in registers).
// h lives in LDS; 2 barriers per step.
// ---------------------------------------------------------------------------
__global__ __launch_bounds__(256) void lstm_scan_kernel(
    const float* __restrict__ xz, const float* __restrict__ W_hh,
    float* __restrict__ hseq)
{
    __shared__ float hsh[HH];
    __shared__ float zsh[256];
    const int j = threadIdx.x;

    float whh[HH];
    #pragma unroll
    for (int k = 0; k < HH; ++k) whh[k] = W_hh[j * HH + k];

    float c = 0.f;
    if (j < HH) hsh[j] = 0.f;
    __syncthreads();

    for (int t = 0; t < TT; ++t) {
        float z = xz[t * 256 + j];
        #pragma unroll
        for (int k = 0; k < HH; ++k) z = fmaf(whh[k], hsh[k], z);
        zsh[j] = z;
        __syncthreads();
        if (j < HH) {
            const float zi = zsh[j];
            const float zf = zsh[j + 64];
            const float zg = zsh[j + 128];
            const float zo = zsh[j + 192];
            const float ig = 1.f / (1.f + __expf(-zi));
            const float fg = 1.f / (1.f + __expf(-zf));
            const float og = 1.f / (1.f + __expf(-zo));
            const float gg = tanhf(zg);
            c = fg * c + ig * gg;
            const float h = og * tanhf(c);
            hsh[j] = h;
            hseq[t * HH + j] = h;
        }
        __syncthreads();
    }
}

// ---------------------------------------------------------------------------
// Kernel B: fused 3-layer MLP with WMMA f16 (accumulate f32).
// 256 threads = 8 waves; each wave owns a 16-row strip; block = 128 rows.
// grid = T*N/128 = 8000 blocks.
//
// CDNA5 16-bit A fragment (16x32): lane m=lane&15, g=lane>>4;
//   halves 0..7  = K in [ks*32 +      g*8, +8)
//   halves 8..15 = K in [ks*32 + 16 + g*8, +8)
// 16-bit B fragment (32x16): column n=lane&15; same K mapping as A.
// f32 C/D fragment: element r -> row r+8*g, column lane&15.
// ---------------------------------------------------------------------------
__global__ __launch_bounds__(256) void sdf_mlp_kernel(
    const float* __restrict__ ind,    // [T*N, DI]
    const int*   __restrict__ masks,  // [T*N]
    const float* __restrict__ hseq,   // [T, H]
    const float* __restrict__ W1, const float* __restrict__ b1,
    const float* __restrict__ W2, const float* __restrict__ b2,
    const float* __restrict__ W3, const float* __restrict__ b3,
    float* __restrict__ outW)         // [T*N]
{
    __shared__ __align__(16) _Float16 sA [128 * FP];   // 32 KB input tile (f16)
    __shared__ __align__(16) _Float16 sW1[ DH_ * FP];  // 16 KB
    __shared__ __align__(16) _Float16 sW2[ DH_ * DH_]; //  8 KB
    __shared__ __align__(16) _Float16 sW3[ DH_];       // 128 B
    __shared__ __align__(16) _Float16 sXa[8 * 16 * DH_]; // 16 KB inter-layer staging
    __shared__ float sB1[DH_], sB2[DH_];

    const int tid = threadIdx.x;
    const int blockBase = blockIdx.x * 128;

    // ---- stage input tile: 2 threads per row, 64 features each, f32->f16 ----
    {
        const int rl = tid >> 1;
        const int fh = (tid & 1) * 64;
        const int row  = blockBase + rl;
        const int trow = row / NN;
        const float* indRow = ind + row * DI_;
        const float* hRow   = hseq + trow * HH;
        #pragma unroll
        for (int cchunk = 0; cchunk < 8; ++cchunk) {
            h8 pk;
            #pragma unroll
            for (int e = 0; e < 8; ++e) {
                const int f = fh + cchunk * 8 + e;
                float v;
                if (f < DI_)            v = indRow[f];
                else if (f < DI_ + HH)  v = hRow[f - DI_];
                else                    v = 0.f;
                pk[e] = (_Float16)v;
            }
            *reinterpret_cast<h8*>(&sA[rl * FP + fh + cchunk * 8]) = pk;
        }
    }
    // ---- stage W1 (pad 110->128) ----
    {
        const int j = tid >> 2, q = tid & 3;
        const float* w1r = W1 + j * (DI_ + HH);
        #pragma unroll
        for (int cchunk = 0; cchunk < 4; ++cchunk) {
            h8 pk;
            #pragma unroll
            for (int e = 0; e < 8; ++e) {
                const int f = q * 32 + cchunk * 8 + e;
                pk[e] = (f < DI_ + HH) ? (_Float16)w1r[f] : (_Float16)0.f;
            }
            *reinterpret_cast<h8*>(&sW1[j * FP + q * 32 + cchunk * 8]) = pk;
        }
    }
    // ---- stage W2 ----
    {
        const int j = tid >> 2, q = tid & 3;
        #pragma unroll
        for (int cchunk = 0; cchunk < 2; ++cchunk) {
            h8 pk;
            #pragma unroll
            for (int e = 0; e < 8; ++e) {
                const int f = q * 16 + cchunk * 8 + e;
                pk[e] = (_Float16)W2[j * DH_ + f];
            }
            *reinterpret_cast<h8*>(&sW2[j * DH_ + q * 16 + cchunk * 8]) = pk;
        }
    }
    if (tid < DH_) {
        sW3[tid] = (_Float16)W3[tid];
        sB1[tid] = b1[tid];
        sB2[tid] = b2[tid];
    }
    __syncthreads();

    // ---- per-wave WMMA pipeline ----
    const int wave = tid >> 5, lane = tid & 31;
    const int m = lane & 15, g = lane >> 4;
    const int rl0 = wave * 16;
    _Float16* myX = &sXa[wave * 16 * DH_];

    // Layer 1: [16 x 128] x [128 x 64]  -> 4 col-groups, 4 k-steps
    v16h aF[4];
    #pragma unroll
    for (int ks = 0; ks < 4; ++ks) {
        const h8 lo = *reinterpret_cast<const h8*>(&sA[(rl0 + m) * FP + ks * 32 + g * 8]);
        const h8 hi = *reinterpret_cast<const h8*>(&sA[(rl0 + m) * FP + ks * 32 + 16 + g * 8]);
        aF[ks] = cat16(lo, hi);
    }
    v8f acc1[4];
    #pragma unroll
    for (int cg = 0; cg < 4; ++cg) acc1[cg] = (v8f)(0.0f);
    #pragma unroll
    for (int cg = 0; cg < 4; ++cg) {
        #pragma unroll
        for (int ks = 0; ks < 4; ++ks) {
            const h8 blo = *reinterpret_cast<const h8*>(&sW1[(cg * 16 + m) * FP + ks * 32 + g * 8]);
            const h8 bhi = *reinterpret_cast<const h8*>(&sW1[(cg * 16 + m) * FP + ks * 32 + 16 + g * 8]);
            acc1[cg] = __builtin_amdgcn_wmma_f32_16x16x32_f16(
                false, aF[ks], false, cat16(blo, bhi), (short)0, acc1[cg], false, false);
        }
    }
    // bias + relu, stage f16 for layer 2
    #pragma unroll
    for (int cg = 0; cg < 4; ++cg) {
        const float bv = sB1[cg * 16 + m];
        #pragma unroll
        for (int r = 0; r < 8; ++r) {
            const float v = fmaxf(acc1[cg][r] + bv, 0.f);
            myX[(r + 8 * g) * DH_ + cg * 16 + m] = (_Float16)v;
        }
    }
    __syncthreads();

    // Layer 2: [16 x 64] x [64 x 64] -> 4 col-groups, 2 k-steps
    v16h a2[2];
    #pragma unroll
    for (int ks = 0; ks < 2; ++ks) {
        const h8 lo = *reinterpret_cast<const h8*>(&myX[m * DH_ + ks * 32 + g * 8]);
        const h8 hi = *reinterpret_cast<const h8*>(&myX[m * DH_ + ks * 32 + 16 + g * 8]);
        a2[ks] = cat16(lo, hi);
    }
    v8f acc2[4];
    #pragma unroll
    for (int cg = 0; cg < 4; ++cg) acc2[cg] = (v8f)(0.0f);
    #pragma unroll
    for (int cg = 0; cg < 4; ++cg) {
        #pragma unroll
        for (int ks = 0; ks < 2; ++ks) {
            const h8 blo = *reinterpret_cast<const h8*>(&sW2[(cg * 16 + m) * DH_ + ks * 32 + g * 8]);
            const h8 bhi = *reinterpret_cast<const h8*>(&sW2[(cg * 16 + m) * DH_ + ks * 32 + 16 + g * 8]);
            acc2[cg] = __builtin_amdgcn_wmma_f32_16x16x32_f16(
                false, a2[ks], false, cat16(blo, bhi), (short)0, acc2[cg], false, false);
        }
    }
    __syncthreads();   // safe to overwrite sXa (a2 fragments are in registers)

    #pragma unroll
    for (int cg = 0; cg < 4; ++cg) {
        const float bv = sB2[cg * 16 + m];
        #pragma unroll
        for (int r = 0; r < 8; ++r) {
            const float v = fmaxf(acc2[cg][r] + bv, 0.f);
            myX[(r + 8 * g) * DH_ + cg * 16 + m] = (_Float16)v;
        }
    }
    __syncthreads();

    // Layer 3: dot with W3 as a WMMA whose B has only column 0 populated
    v8f acc3 = (v8f)(0.0f);
    #pragma unroll
    for (int ks = 0; ks < 2; ++ks) {
        const h8 lo = *reinterpret_cast<const h8*>(&myX[m * DH_ + ks * 32 + g * 8]);
        const h8 hi = *reinterpret_cast<const h8*>(&myX[m * DH_ + ks * 32 + 16 + g * 8]);
        const v16h a3 = cat16(lo, hi);
        h8 blo = (h8)((_Float16)0.f);
        h8 bhi = (h8)((_Float16)0.f);
        if (m == 0) {
            blo = *reinterpret_cast<const h8*>(&sW3[ks * 32 + g * 8]);
            bhi = *reinterpret_cast<const h8*>(&sW3[ks * 32 + 16 + g * 8]);
        }
        acc3 = __builtin_amdgcn_wmma_f32_16x16x32_f16(
            false, a3, false, cat16(blo, bhi), (short)0, acc3, false, false);
    }

    // column 0 of D holds the per-row weight: lanes 0 (rows 0..7) and 16 (rows 8..15)
    const float b3v = b3[0];
    if (m == 0) {
        #pragma unroll
        for (int r = 0; r < 8; ++r) {
            const int grow = blockBase + rl0 + 8 * g + r;
            const float wv = acc3[r] + b3v;
            outW[grow] = masks[grow] ? wv : 0.f;
        }
    }
}

// ---------------------------------------------------------------------------
// Kernel C1: per-timestep deterministic reduction.
// grid = T blocks; block t computes dot[t] = sum_n ret*w, cnt[t] = sum_n mask.
// ---------------------------------------------------------------------------
__global__ __launch_bounds__(256) void reduce_t_kernel(
    const int* __restrict__ masks, const float* __restrict__ ret,
    const float* __restrict__ w, float* __restrict__ dots, float* __restrict__ cnts)
{
    __shared__ float ss[256], sc[256];
    const int tid = threadIdx.x;
    const int base = blockIdx.x * NN;
    float s = 0.f, c = 0.f;
    for (int n = tid; n < NN; n += 256) {
        const int idx = base + n;
        c += (float)masks[idx];
        s += ret[idx] * w[idx];
    }
    ss[tid] = s; sc[tid] = c;
    __syncthreads();
    #pragma unroll
    for (int off = 128; off > 0; off >>= 1) {
        if (tid < off) { ss[tid] += ss[tid + off]; sc[tid] += sc[tid + off]; }
        __syncthreads();
    }
    if (tid == 0) { dots[blockIdx.x] = ss[0]; cnts[blockIdx.x] = sc[0]; }
}

// ---------------------------------------------------------------------------
// Kernel C2: sdf[t] = dot[t]/cnt[t] * mean(cnt) + 1.  One block, 256 threads.
// ---------------------------------------------------------------------------
__global__ __launch_bounds__(256) void finalize_kernel(
    const float* __restrict__ dots, const float* __restrict__ cnts,
    float* __restrict__ sdf)
{
    __shared__ float sc[256];
    const int tid = threadIdx.x;
    sc[tid] = cnts[tid];
    __syncthreads();
    #pragma unroll
    for (int off = 128; off > 0; off >>= 1) {
        if (tid < off) sc[tid] += sc[tid + off];
        __syncthreads();
    }
    const float meanCnt = sc[0] * (1.f / 256.f);
    sdf[tid] = dots[tid] / cnts[tid] * meanCnt + 1.f;
}

// ---------------------------------------------------------------------------
// Launch: inputs in setup_inputs() order.
// d_out = [ sdf (256 f32) | weights (1,024,000 f32) ].
// ---------------------------------------------------------------------------
extern "C" void kernel_launch(void* const* d_in, const int* in_sizes, int n_in,
                              void* d_out, int out_size, void* d_ws, size_t ws_size,
                              hipStream_t stream) {
    (void)in_sizes; (void)n_in; (void)out_size; (void)ws_size;

    const float* macro = (const float*)d_in[0];
    const float* ind   = (const float*)d_in[1];
    const int*   masks = (const int*)  d_in[2];   // bool mask -> int per harness convention
    const float* ret   = (const float*)d_in[3];
    const float* W_ih  = (const float*)d_in[4];
    const float* W_hh  = (const float*)d_in[5];
    const float* b_ih  = (const float*)d_in[6];
    const float* b_hh  = (const float*)d_in[7];
    const float* W1    = (const float*)d_in[8];
    const float* b1    = (const float*)d_in[9];
    const float* W2    = (const float*)d_in[10];
    const float* b2    = (const float*)d_in[11];
    const float* W3    = (const float*)d_in[12];
    const float* b3    = (const float*)d_in[13];

    char* ws = (char*)d_ws;
    float* ws_xz   = (float*)(ws);                       // T*256 f32 = 256 KB
    float* ws_h    = (float*)(ws + 262144);              // T*64  f32 =  64 KB
    float* ws_dots = (float*)(ws + 262144 + 65536);      // T f32
    float* ws_cnts = ws_dots + TT;                       // T f32

    float* sdf  = (float*)d_out;
    float* outW = sdf + TT;

    lstm_xz_kernel  <<<TT, 256, 0, stream>>>(macro, W_ih, b_ih, b_hh, ws_xz);
    lstm_scan_kernel<<<1,  256, 0, stream>>>(ws_xz, W_hh, ws_h);
    sdf_mlp_kernel  <<<(TT * NN) / 128, 256, 0, stream>>>(
        ind, masks, ws_h, W1, b1, W2, b2, W3, b3, outW);
    reduce_t_kernel <<<TT, 256, 0, stream>>>(masks, ret, outW, ws_dots, ws_cnts);
    finalize_kernel <<<1,  256, 0, stream>>>(ws_dots, ws_cnts, sdf);
}
